// ResidualAttentionBlock_71047349011048
// MI455X (gfx1250) — matrix operand order repaired and checked by reading, not compiled, not run on hardware
//
#include <hip/hip_runtime.h>
#include <hip/hip_bf16.h>

typedef __attribute__((ext_vector_type(16))) _Float16 v16h;
typedef __attribute__((ext_vector_type(8)))  float    v8f;

// Problem constants (from reference setup_inputs)
constexpr int BS = 64, T = 256, D = 128, H = 4, DH = 128;
constexpr int TOK = BS * T;            // 16384
constexpr int NQV = 3 * H * DH;        // 1536
constexpr int RED_BLK = 1024;

// ---- workspace layout (bytes) ----
constexpr size_t OFF_STATS = 0;                                   // 2 floats (mean, inv_std)
constexpr size_t OFF_PART  = 256;                                 // RED_BLK*2 floats
constexpr size_t OFF_XN    = OFF_PART + (size_t)RED_BLK * 2 * 4;  // TOK*D f32
constexpr size_t OFF_KL1   = OFF_XN + (size_t)TOK * D * 4;        // TOK*H f32
constexpr size_t OFF_Q     = OFF_KL1 + (size_t)TOK * H * 4;       // TOK*(H*DH) f32 (wk-scaled)
constexpr size_t OFF_QL1   = OFF_Q + (size_t)TOK * H * DH * 4;    // TOK*H f32
constexpr size_t OFF_VF    = OFF_QL1 + (size_t)TOK * H * 4;       // [b][h][w][s] f16 (transposed!)
constexpr size_t OFF_VB    = OFF_VF + (size_t)TOK * H * DH * 2;   // [b][h][w][d] f16 (transposed!)
constexpr size_t OFF_A     = OFF_VB + (size_t)TOK * H * DH * 2;   // [b][h][s][d] f32 scores
constexpr size_t OFF_AF    = OFF_A + (size_t)BS * H * T * T * 4;  // [b][h][d][s] f16
constexpr size_t OFF_AB    = OFF_AF + (size_t)BS * H * T * T * 2; // [b][h][s][d] f16
constexpr size_t OFF_WQH   = OFF_AB + (size_t)BS * H * T * T * 2; // wqv_w f16 [n][k]
constexpr size_t OFF_FWH   = OFF_WQH + (size_t)NQV * D * 2;       // fanin_w f16 [n][k]
constexpr size_t OFF_XNH   = OFF_FWH + (size_t)D * D * 2;         // xn f16
// aliases (dead buffers reused):
constexpr size_t OFF_BF    = OFF_A;                               // [b][h][d][w] f32 (scores dead)
constexpr size_t OFF_BB    = OFF_A + (size_t)BS * H * T * DH * 4; // [b][h][s][w] f32
constexpr size_t OFF_Y0G   = OFF_XN;                              // TOK*D f16 (xn f32 dead)

// ---------------- WMMA fragment helpers (CDNA5 16x16x32 f16 layouts) ----------------
// A (16x32, M=lane&15): halves 0..7 = K[kba..kba+7], halves 8..15 = K[16+kba..16+kba+7]
__device__ __forceinline__ v16h load_a_frag(const _Float16* __restrict__ Arow, int k0, int kba) {
  v16h r;
#pragma unroll
  for (int i = 0; i < 8; ++i) {
    r[i]     = Arow[k0 + kba + i];
    r[8 + i] = Arow[k0 + 16 + kba + i];
  }
  return r;
}
// B (32x16, N=lane&15), supplied as B^T rows (contiguous in K): halves 0..7 = K[kbb/2..kbb/2+7], halves 8..15 = K[16+kbb/2..16+kbb/2+7]
__device__ __forceinline__ v16h load_b_frag(const _Float16* __restrict__ Bcol, int k0, int kbb) {
  v16h r;
#pragma unroll
  for (int i = 0; i < 8; ++i) {
    r[i]     = Bcol[k0 + (kbb >> 1) + i];
    r[8 + i] = Bcol[k0 + 16 + (kbb >> 1) + i];
  }
  return r;
}

// ---------------- Stage 0: deterministic global sum / sumsq ----------------
__global__ void k_reduce(const float* __restrict__ x, float* __restrict__ part, int n) {
  __shared__ float s1[256], s2[256];
  int tid = threadIdx.x;
  float a = 0.f, b = 0.f;
  for (int i = blockIdx.x * 256 + tid; i < n; i += gridDim.x * 256) {
    float v = x[i]; a += v; b += v * v;
  }
  s1[tid] = a; s2[tid] = b; __syncthreads();
  for (int off = 128; off > 0; off >>= 1) {
    if (tid < off) { s1[tid] += s1[tid + off]; s2[tid] += s2[tid + off]; }
    __syncthreads();
  }
  if (tid == 0) { part[blockIdx.x * 2] = s1[0]; part[blockIdx.x * 2 + 1] = s2[0]; }
}

__global__ void k_finalize(const float* __restrict__ part, float* __restrict__ stats, float n) {
  __shared__ float s1[RED_BLK], s2[RED_BLK];
  int tid = threadIdx.x;
  s1[tid] = part[tid * 2]; s2[tid] = part[tid * 2 + 1];
  __syncthreads();
  for (int off = RED_BLK / 2; off > 0; off >>= 1) {
    if (tid < off) { s1[tid] += s1[tid + off]; s2[tid] += s2[tid + off]; }
    __syncthreads();
  }
  if (tid == 0) {
    float mean = s1[0] / n;
    float var  = (s2[0] - s1[0] * s1[0] / n) / (n - 1.0f);  // ddof=1
    stats[0] = mean;
    stats[1] = rsqrtf(var);
  }
}

// ---------------- weight / activation f32 -> f16 conversion ----------------
__global__ void k_cvt(const float* __restrict__ src, _Float16* __restrict__ dst, int n) {
  int i = blockIdx.x * 256 + threadIdx.x;
  if (i < n) dst[i] = (_Float16)src[i];
}

// ---------------- Stage 1: xn (f32 + f16) + per-head k_l1 ----------------
__global__ void k_xn(const float* __restrict__ x, const float* __restrict__ wk,
                     const float* __restrict__ stats, float* __restrict__ xn,
                     _Float16* __restrict__ xnh, float* __restrict__ k_l1) {
  int t = blockIdx.x, w = threadIdx.x;  // block 128
  float mean = stats[0], inv = stats[1];
  float v = (x[(size_t)t * D + w] - mean) * inv;
  xn[(size_t)t * D + w] = v;
  xnh[(size_t)t * D + w] = (_Float16)v;
  __shared__ float s[128];
  for (int h = 0; h < H; ++h) {
    s[w] = fabsf(v * wk[h * D + w]);
    __syncthreads();
    for (int off = 64; off > 0; off >>= 1) {
      if (w < off) s[w] += s[w + off];
      __syncthreads();
    }
    if (w == 0) k_l1[(size_t)t * H + h] = s[0];
    __syncthreads();
  }
}

// ---------------- Stage 2: QV projection GEMM (16x64 strip / wave) ----------------
// C[16384 x 1536] = xnh @ wqh^T + b ; scatter q (wk-scaled, f32), vfT/vbT (f16, [b][h][w][s])
__global__ void k_qv(const _Float16* __restrict__ xnh, const _Float16* __restrict__ wqh,
                     const float* __restrict__ wqv_b, const float* __restrict__ wk,
                     float* __restrict__ q, _Float16* __restrict__ vfT,
                     _Float16* __restrict__ vbT) {
  constexpr int STRIPS = NQV / 64;  // 24
  int wg = blockIdx.x * 8 + (threadIdx.x >> 5);
  int ts = wg % STRIPS, tm = wg / STRIPS;
  int lane = threadIdx.x & 31;
  int nn = lane & 15;
  int kba = (lane >> 4) << 3;   // A: 0 or 8
  int kbb = (lane >> 4) << 4;   // B: 0 or 16
  int m0 = tm * 16, n0 = ts * 64;
  const _Float16* Arow = xnh + (size_t)(m0 + nn) * D;
  const _Float16* Bc   = wqh + (size_t)(n0 + nn) * D;  // B^T row n (contiguous in k)
  v8f acc0 = {}, acc1 = {}, acc2 = {}, acc3 = {};
  for (int k0 = 0; k0 < D; k0 += 32) {
    v16h av = load_a_frag(Arow, k0, kba);
    acc0 = __builtin_amdgcn_wmma_f32_16x16x32_f16(false, av, false, load_b_frag(Bc,              k0, kbb), (short)0, acc0, false, false);
    acc1 = __builtin_amdgcn_wmma_f32_16x16x32_f16(false, av, false, load_b_frag(Bc + 16 * D,     k0, kbb), (short)0, acc1, false, false);
    acc2 = __builtin_amdgcn_wmma_f32_16x16x32_f16(false, av, false, load_b_frag(Bc + 32 * D,     k0, kbb), (short)0, acc2, false, false);
    acc3 = __builtin_amdgcn_wmma_f32_16x16x32_f16(false, av, false, load_b_frag(Bc + 48 * D,     k0, kbb), (short)0, acc3, false, false);
  }
  int mb = m0 + ((lane >> 4) << 3);
#pragma unroll
  for (int sub = 0; sub < 4; ++sub) {
    v8f acc = (sub == 0) ? acc0 : (sub == 1) ? acc1 : (sub == 2) ? acc2 : acc3;
    int col = n0 + sub * 16 + nn;
    float bias = wqv_b[col];
    float wkq = (col < H * DH) ? wk[col] : 0.f;
#pragma unroll
    for (int r = 0; r < 8; ++r) {
      int row = mb + r;
      float val = acc[r] + bias;
      int b = row >> 8, s = row & (T - 1);
      if (col < H * DH) {
        q[(size_t)row * (H * DH) + col] = val * wkq;
      } else if (col < 2 * H * DH) {
        int c = col - H * DH; int h = c >> 7; int w = c & 127;
        vfT[(((size_t)(b * H + h) * DH) + w) * T + s] = (_Float16)val;
      } else {
        int c = col - 2 * H * DH; int h = c >> 7; int w = c & 127;
        vbT[(((size_t)(b * H + h) * DH) + w) * T + s] = (_Float16)val;
      }
    }
  }
}

// ---------------- Stage 3: q_l1 ----------------
__global__ void k_ql1(const float* __restrict__ q, float* __restrict__ q_l1) {
  int t = blockIdx.x, i = threadIdx.x;  // block 512
  __shared__ float s[512];
  s[i] = fabsf(q[(size_t)t * (H * DH) + i]);
  __syncthreads();
  for (int off = 64; off > 0; off >>= 1) {
    if ((i & 127) < off) s[i] += s[i + off];
    __syncthreads();
  }
  if ((i & 127) == 0) q_l1[(size_t)t * H + (i >> 7)] = s[i];
}

// ---------------- Stage 4: L1 scores (VALU, LDS-tiled) ----------------
__global__ void k_scores(const float* __restrict__ q, const float* __restrict__ xn,
                         const float* __restrict__ wk, const float* __restrict__ q_l1,
                         const float* __restrict__ k_l1, float* __restrict__ a) {
  int blk = blockIdx.x;
  int dt = blk & 15, st = (blk >> 4) & 15, h = (blk >> 8) & 3, b = blk >> 10;
  int tid = threadIdx.x;  // 256
  __shared__ float xs[16][128];
  __shared__ float qs[16][128];
  __shared__ float wks[128];
  int s0 = st * 16, d0 = dt * 16;
  for (int i = tid; i < 16 * 128; i += 256) {
    int r = i >> 7, c = i & 127;
    xs[r][c] = xn[((size_t)(b * T + s0 + r)) * D + c];
    qs[r][c] = q[((size_t)(b * T + d0 + r)) * (H * DH) + h * DH + c];
  }
  if (tid < 128) wks[tid] = wk[h * D + tid];
  __syncthreads();
  int dl = tid & 15, sl = tid >> 4;
  float l1 = 0.f;
#pragma unroll 8
  for (int w = 0; w < D; ++w) l1 += fabsf(qs[dl][w] - xs[sl][w] * wks[w]);
  int sg = s0 + sl, dg = d0 + dl;
  float ac = 0.7071f * (q_l1[(size_t)(b * T + dg) * H + h] + k_l1[(size_t)(b * T + sg) * H + h]);
  a[(((size_t)(b * H + h) * T) + sg) * T + dg] = (-l1 - ac) * 0.1f;
}

// ---------------- Stage 5: softmaxes (wave per row/col) ----------------
__global__ void k_softmax_f(const float* __restrict__ a, _Float16* __restrict__ af) {
  int wid = blockIdx.x * 8 + (threadIdx.x >> 5);
  int lane = threadIdx.x & 31;
  int d = wid & (T - 1); int bh = wid >> 8;
  const float* base = a + ((size_t)bh * T) * T + d;  // column: base[s*T]
  float v[8]; float mx = -3.4e38f;
#pragma unroll
  for (int i = 0; i < 8; ++i) { v[i] = base[(size_t)(lane + i * 32) * T]; mx = fmaxf(mx, v[i]); }
#pragma unroll
  for (int off = 16; off > 0; off >>= 1) mx = fmaxf(mx, __shfl_xor(mx, off));
  float sum = 0.f;
#pragma unroll
  for (int i = 0; i < 8; ++i) { v[i] = __expf(v[i] - mx); sum += v[i]; }
#pragma unroll
  for (int off = 16; off > 0; off >>= 1) sum += __shfl_xor(sum, off);
  float r = 1.f / sum;
  _Float16* dst = af + ((size_t)bh * T + d) * T;  // af[b][h][d][s]
#pragma unroll
  for (int i = 0; i < 8; ++i) dst[lane + i * 32] = (_Float16)(v[i] * r);
}

__global__ void k_softmax_b(const float* __restrict__ a, _Float16* __restrict__ ab) {
  int wid = blockIdx.x * 8 + (threadIdx.x >> 5);
  int lane = threadIdx.x & 31;
  int s = wid & (T - 1); int bh = wid >> 8;
  const float* base = a + ((size_t)bh * T + s) * T;  // row: base[d]
  float v[8]; float mx = -3.4e38f;
#pragma unroll
  for (int i = 0; i < 8; ++i) { v[i] = base[lane + i * 32]; mx = fmaxf(mx, v[i]); }
#pragma unroll
  for (int off = 16; off > 0; off >>= 1) mx = fmaxf(mx, __shfl_xor(mx, off));
  float sum = 0.f;
#pragma unroll
  for (int i = 0; i < 8; ++i) { v[i] = __expf(v[i] - mx); sum += v[i]; }
#pragma unroll
  for (int off = 16; off > 0; off >>= 1) sum += __shfl_xor(sum, off);
  float r = 1.f / sum;
  _Float16* dst = ab + ((size_t)bh * T + s) * T;  // ab[b][h][s][d]
#pragma unroll
  for (int i = 0; i < 8; ++i) dst[lane + i * 32] = (_Float16)(v[i] * r);
}

// ---------------- Stage 6: batched f16 GEMM, B given transposed (16x64 / wave) ----------------
// C[M x N] = A[M x K](row major) @ B, with B supplied as B^T[N x K] row major. f32 out.
__global__ void k_gemm_f16t(const _Float16* __restrict__ A, const _Float16* __restrict__ BT,
                            float* __restrict__ C, int M, int N, int K,
                            long sA, long sB, long sC) {
  int wg = blockIdx.x * 8 + (threadIdx.x >> 5);
  int strips = N >> 6, tilesM = M >> 4;
  int per = tilesM * strips;
  int batch = wg / per;
  int tt = wg % per;
  int tm = tt / strips, ts = tt % strips;
  int lane = threadIdx.x & 31;
  int nn = lane & 15;
  int kba = (lane >> 4) << 3;
  int kbb = (lane >> 4) << 4;
  int m0 = tm * 16, n0 = ts * 64;
  const _Float16* Arow = A + (size_t)batch * sA + (size_t)(m0 + nn) * K;
  const _Float16* Bc   = BT + (size_t)batch * sB + (size_t)(n0 + nn) * K;
  v8f acc0 = {}, acc1 = {}, acc2 = {}, acc3 = {};
  for (int k0 = 0; k0 < K; k0 += 32) {
    v16h av = load_a_frag(Arow, k0, kba);
    acc0 = __builtin_amdgcn_wmma_f32_16x16x32_f16(false, av, false, load_b_frag(Bc,                    k0, kbb), (short)0, acc0, false, false);
    acc1 = __builtin_amdgcn_wmma_f32_16x16x32_f16(false, av, false, load_b_frag(Bc + (size_t)16 * K,   k0, kbb), (short)0, acc1, false, false);
    acc2 = __builtin_amdgcn_wmma_f32_16x16x32_f16(false, av, false, load_b_frag(Bc + (size_t)32 * K,   k0, kbb), (short)0, acc2, false, false);
    acc3 = __builtin_amdgcn_wmma_f32_16x16x32_f16(false, av, false, load_b_frag(Bc + (size_t)48 * K,   k0, kbb), (short)0, acc3, false, false);
  }
  float* Cb = C + (size_t)batch * sC;
  int mb = m0 + ((lane >> 4) << 3);
#pragma unroll
  for (int sub = 0; sub < 4; ++sub) {
    v8f acc = (sub == 0) ? acc0 : (sub == 1) ? acc1 : (sub == 2) ? acc2 : acc3;
    int n = n0 + sub * 16 + nn;
#pragma unroll
    for (int r = 0; r < 8; ++r) Cb[(size_t)(mb + r) * N + n] = acc[r];
  }
}

// ---------------- Stage 7: head-sum + quick-GELU ----------------
__global__ void k_combine(const float* __restrict__ bf, const float* __restrict__ bb,
                          _Float16* __restrict__ y0g) {
  int t = blockIdx.x, w = threadIdx.x;  // block 128
  int b = t >> 8, s = t & (T - 1);
  float acc = 0.f;
#pragma unroll
  for (int h = 0; h < H; ++h) {
    size_t idx = (((size_t)(b * H + h) * T) + s) * DH + w;
    acc += bf[idx] + bb[idx];
  }
  float g = acc / (1.f + __expf(-1.702f * acc));
  y0g[(size_t)t * D + w] = (_Float16)g;
}

// ---------------- Stage 8: fanin GEMM + residual (16x64 strip / wave) ----------------
__global__ void k_fanin(const _Float16* __restrict__ y0g, const _Float16* __restrict__ fwh,
                        const float* __restrict__ fb, const float* __restrict__ x,
                        float* __restrict__ out) {
  int wg = blockIdx.x * 8 + (threadIdx.x >> 5);
  int ts = wg & 1, tm = wg >> 1;  // N=128 -> 2 strips of 64
  int lane = threadIdx.x & 31;
  int nn = lane & 15;
  int kba = (lane >> 4) << 3;
  int kbb = (lane >> 4) << 4;
  int m0 = tm * 16, n0 = ts * 64;
  const _Float16* Arow = y0g + (size_t)(m0 + nn) * D;
  const _Float16* Bc   = fwh + (size_t)(n0 + nn) * D;  // fanin_w[n][k] = B^T
  v8f acc0 = {}, acc1 = {}, acc2 = {}, acc3 = {};
  for (int k0 = 0; k0 < D; k0 += 32) {
    v16h av = load_a_frag(Arow, k0, kba);
    acc0 = __builtin_amdgcn_wmma_f32_16x16x32_f16(false, av, false, load_b_frag(Bc,          k0, kbb), (short)0, acc0, false, false);
    acc1 = __builtin_amdgcn_wmma_f32_16x16x32_f16(false, av, false, load_b_frag(Bc + 16 * D, k0, kbb), (short)0, acc1, false, false);
    acc2 = __builtin_amdgcn_wmma_f32_16x16x32_f16(false, av, false, load_b_frag(Bc + 32 * D, k0, kbb), (short)0, acc2, false, false);
    acc3 = __builtin_amdgcn_wmma_f32_16x16x32_f16(false, av, false, load_b_frag(Bc + 48 * D, k0, kbb), (short)0, acc3, false, false);
  }
  int mb = m0 + ((lane >> 4) << 3);
#pragma unroll
  for (int sub = 0; sub < 4; ++sub) {
    v8f acc = (sub == 0) ? acc0 : (sub == 1) ? acc1 : (sub == 2) ? acc2 : acc3;
    int n = n0 + sub * 16 + nn;
    float bias = fb[n];
#pragma unroll
    for (int r = 0; r < 8; ++r) {
      int m = mb + r;
      out[(size_t)m * D + n] = x[(size_t)m * D + n] + acc[r] + bias;
    }
  }
}

extern "C" void kernel_launch(void* const* d_in, const int* in_sizes, int n_in,
                              void* d_out, int out_size, void* d_ws, size_t ws_size,
                              hipStream_t stream) {
  const float* x       = (const float*)d_in[0];
  const float* wk      = (const float*)d_in[1];
  const float* wqv_w   = (const float*)d_in[2];
  const float* wqv_b   = (const float*)d_in[3];
  const float* fanin_w = (const float*)d_in[4];
  const float* fanin_b = (const float*)d_in[5];
  float* out = (float*)d_out;

  char* ws = (char*)d_ws;
  float*    stats = (float*)(ws + OFF_STATS);
  float*    part  = (float*)(ws + OFF_PART);
  float*    xn    = (float*)(ws + OFF_XN);
  float*    k_l1  = (float*)(ws + OFF_KL1);
  float*    q     = (float*)(ws + OFF_Q);
  float*    q_l1  = (float*)(ws + OFF_QL1);
  _Float16* vfT   = (_Float16*)(ws + OFF_VF);
  _Float16* vbT   = (_Float16*)(ws + OFF_VB);
  float*    sc    = (float*)(ws + OFF_A);
  _Float16* af    = (_Float16*)(ws + OFF_AF);
  _Float16* ab    = (_Float16*)(ws + OFF_AB);
  _Float16* wqh   = (_Float16*)(ws + OFF_WQH);
  _Float16* fwh   = (_Float16*)(ws + OFF_FWH);
  _Float16* xnh   = (_Float16*)(ws + OFF_XNH);
  float*    bfp   = (float*)(ws + OFF_BF);
  float*    bbp   = (float*)(ws + OFF_BB);
  _Float16* y0g   = (_Float16*)(ws + OFF_Y0G);

  // weight conversions (independent of stats)
  k_cvt<<<(NQV * D + 255) / 256, 256, 0, stream>>>(wqv_w, wqh, NQV * D);
  k_cvt<<<(D * D + 255) / 256, 256, 0, stream>>>(fanin_w, fwh, D * D);
  // 0: global mean/std (deterministic two-stage tree)
  k_reduce<<<RED_BLK, 256, 0, stream>>>(x, part, TOK * D);
  k_finalize<<<1, RED_BLK, 0, stream>>>(part, stats, (float)(TOK * D));
  // 1: xn + k_l1
  k_xn<<<TOK, 128, 0, stream>>>(x, wk, stats, xn, xnh, k_l1);
  // 2: QV projection GEMM (WMMA, 16x64 strips): 1024*24 waves
  k_qv<<<(TOK / 16) * (NQV / 64) / 8, 256, 0, stream>>>(xnh, wqh, wqv_b, wk, q, vfT, vbT);
  // 3: q_l1
  k_ql1<<<TOK, 512, 0, stream>>>(q, q_l1);
  // 4: L1 scores
  k_scores<<<BS * H * 16 * 16, 256, 0, stream>>>(q, xn, wk, q_l1, k_l1, sc);
  // 5: softmaxes
  k_softmax_f<<<BS * H * T / 8, 256, 0, stream>>>(sc, af);
  k_softmax_b<<<BS * H * T / 8, 256, 0, stream>>>(sc, ab);
  // 6: attention apply GEMMs (WMMA), batch = BS*H = 256, each M=256,N=128,K=256
  {
    int waves = BS * H * (T / 16) * (DH / 64);
    k_gemm_f16t<<<waves / 8, 256, 0, stream>>>(af, vfT, bfp, T, DH, T,
                                               (long)T * T, (long)DH * T, (long)T * DH);
    k_gemm_f16t<<<waves / 8, 256, 0, stream>>>(ab, vbT, bbp, T, DH, T,
                                               (long)T * T, (long)DH * T, (long)T * DH);
  }
  // 7: head sum + quick-GELU
  k_combine<<<TOK, 128, 0, stream>>>(bfp, bbp, y0g);
  // 8: fanin GEMM + residual (16x64 strips)
  k_fanin<<<(TOK / 16) * (D / 64) / 8, 256, 0, stream>>>(y0g, fwh, fanin_b, x, out);

  (void)in_sizes; (void)n_in; (void)out_size; (void)ws_size;
}
